// LoRACrossAttnProcessor_19327352832372
// MI455X (gfx1250) — matrix operand order repaired and checked by reading, not compiled, not run on hardware
//
#include <hip/hip_runtime.h>
#include <hip/hip_bf16.h>

#ifndef __has_builtin
#define __has_builtin(x) 0
#endif
#if __has_builtin(__builtin_amdgcn_global_load_async_to_lds_b128) && \
    __has_builtin(__builtin_amdgcn_s_wait_asynccnt)
#define HAVE_ASYNC_LDS 1
#else
#define HAVE_ASYNC_LDS 0
#endif

typedef __bf16 bf16_t;
typedef __attribute__((ext_vector_type(8)))  __bf16 v8bf;
typedef __attribute__((ext_vector_type(16))) __bf16 v16bf;
typedef __attribute__((ext_vector_type(8)))  float  v8f;
typedef __attribute__((ext_vector_type(4)))  int    v4i;

union BF16x16 { v16bf v; v8bf h[2]; };

static __device__ __forceinline__ bf16_t to_bf16(float x) {
  unsigned u = __builtin_bit_cast(unsigned, x);
  unsigned r = (u + 0x7FFFu + ((u >> 16) & 1u)) >> 16;
  unsigned short s = (unsigned short)r;
  return __builtin_bit_cast(bf16_t, s);
}

static __device__ __forceinline__ v8f wmma_bf16(const BF16x16& a, const BF16x16& b, v8f c) {
  return __builtin_amdgcn_wmma_f32_16x16x32_bf16(false, a.v, false, b.v, (short)0, c, false, false);
}

// 16-byte Global -> LDS copy: async (no VGPR round-trip, ASYNCcnt-tracked) if available.
static __device__ __forceinline__ void async_cp16(const bf16_t* g, bf16_t* l) {
#if HAVE_ASYNC_LDS
  typedef __attribute__((address_space(1))) v4i* gv4i_p;
  typedef __attribute__((address_space(3))) v4i* sv4i_p;
  __builtin_amdgcn_global_load_async_to_lds_b128((gv4i_p)g, (sv4i_p)l, 0, 0);
#else
  (void)g; (void)l;
#endif
}

// ---------------- fp32 -> bf16 conversion ----------------
__global__ void cvt_f32_bf16(const float* __restrict__ in, bf16_t* __restrict__ out, int n4) {
  int i = blockIdx.x * blockDim.x + threadIdx.x;
  if (i >= n4) return;
  float4 f = ((const float4*)in)[i];
  bf16_t* o = out + (size_t)i * 4;
  o[0] = to_bf16(f.x); o[1] = to_bf16(f.y); o[2] = to_bf16(f.z); o[3] = to_bf16(f.w);
}

// -------- fold LoRA into weight, emit transposed bf16: Wt[n][k] = W[k][n] + sum_r down[k][r]*up[r][n]
__global__ void fold_w(const float* __restrict__ w, const float* __restrict__ down,
                       const float* __restrict__ up, bf16_t* __restrict__ wt,
                       int Kd, int N) {
  int idx = blockIdx.x * blockDim.x + threadIdx.x;
  if (idx >= Kd * N) return;
  int n = idx / Kd, k = idx - n * Kd;
  float acc = w[(size_t)k * N + n];
#pragma unroll
  for (int r = 0; r < 8; ++r) acc += down[k * 8 + r] * up[r * N + n];
  wt[idx] = to_bf16(acc);
}

// ---------------- tiled WMMA GEMM: C[M,N] = A[M,K](bf16) * Wt[N,K]^T(bf16) ----------------
// Double-buffered LDS ping-pong; async Global->LDS staging when the toolchain exposes it.
template <bool OUT_F32>
__global__ __launch_bounds__(256) void gemm_bf16(
    const bf16_t* __restrict__ A, const bf16_t* __restrict__ Wt,
    const float* __restrict__ bias, void* __restrict__ Cv,
    int M, int N, int K) {
  __shared__ bf16_t sA[2][128 * 32];
  __shared__ bf16_t sB[2][128 * 32];   // stored as [n][k]
  const int tid = threadIdx.x;
  const int lane = tid & 31, wid = tid >> 5;
  const int waveM = (wid & 3) * 32;   // 4 wave rows x 32
  const int waveN = (wid >> 2) * 64;  // 2 wave cols x 64
  const int m0 = blockIdx.y * 128, n0 = blockIdx.x * 128;

  v8f acc[2][4];
  const v8f zf = {0.f, 0.f, 0.f, 0.f, 0.f, 0.f, 0.f, 0.f};
#pragma unroll
  for (int i = 0; i < 2; i++)
#pragma unroll
    for (int j = 0; j < 4; j++) acc[i][j] = zf;

  const int sRow = tid >> 1, sKp = (tid & 1) * 16;
  const int sOff = sRow * 32 + sKp;
  const bf16_t* gA = A + (size_t)(m0 + sRow) * K + sKp;
  const bf16_t* gB = Wt + (size_t)(n0 + sRow) * K + sKp;
  const int nk = K >> 5;

#if HAVE_ASYNC_LDS
  async_cp16(gA,     &sA[0][sOff]);
  async_cp16(gA + 8, &sA[0][sOff + 8]);
  async_cp16(gB,     &sB[0][sOff]);
  async_cp16(gB + 8, &sB[0][sOff + 8]);
#else
  v8bf ra0 = *(const v8bf*)(gA);
  v8bf ra1 = *(const v8bf*)(gA + 8);
  v8bf rb0 = *(const v8bf*)(gB);
  v8bf rb1 = *(const v8bf*)(gB + 8);
#endif

  for (int i = 0; i < nk; ++i) {
    const int buf = i & 1;
#if HAVE_ASYNC_LDS
    __builtin_amdgcn_s_wait_asynccnt(0);
    __syncthreads();
    if (i + 1 < nk) {
      const int kb = (i + 1) << 5;
      async_cp16(gA + kb,     &sA[buf ^ 1][sOff]);
      async_cp16(gA + kb + 8, &sA[buf ^ 1][sOff + 8]);
      async_cp16(gB + kb,     &sB[buf ^ 1][sOff]);
      async_cp16(gB + kb + 8, &sB[buf ^ 1][sOff + 8]);
    }
#else
    *(v8bf*)&sA[buf][sOff]     = ra0;
    *(v8bf*)&sA[buf][sOff + 8] = ra1;
    *(v8bf*)&sB[buf][sOff]     = rb0;
    *(v8bf*)&sB[buf][sOff + 8] = rb1;
    __syncthreads();
    if (i + 1 < nk) {
      const int kb = (i + 1) << 5;
      ra0 = *(const v8bf*)(gA + kb);
      ra1 = *(const v8bf*)(gA + kb + 8);
      rb0 = *(const v8bf*)(gB + kb);
      rb1 = *(const v8bf*)(gB + kb + 8);
    }
#endif

    BF16x16 aF[2], bF[4];
    const int kb0a = (lane < 16) ? 0 : 8;   // A frag: lane-half K split {0-7,16-23}/{8-15,24-31}
    const int kb0b = (lane < 16) ? 0 : 8;   // B frag: lane-half K split {0-7,16-23}/{8-15,24-31}
#pragma unroll
    for (int mt = 0; mt < 2; mt++) {
      const bf16_t* p = &sA[buf][(waveM + mt * 16 + (lane & 15)) * 32 + kb0a];
      aF[mt].h[0] = *(const v8bf*)p;
      aF[mt].h[1] = *(const v8bf*)(p + 16);
    }
#pragma unroll
    for (int nt = 0; nt < 4; nt++) {
      const bf16_t* p = &sB[buf][(waveN + nt * 16 + (lane & 15)) * 32 + kb0b];
      bF[nt].h[0] = *(const v8bf*)p;
      bF[nt].h[1] = *(const v8bf*)(p + 16);
    }
#pragma unroll
    for (int mt = 0; mt < 2; mt++)
#pragma unroll
      for (int nt = 0; nt < 4; nt++)
        acc[mt][nt] = wmma_bf16(aF[mt], bF[nt], acc[mt][nt]);
  }

#pragma unroll
  for (int mt = 0; mt < 2; mt++)
#pragma unroll
    for (int nt = 0; nt < 4; nt++)
#pragma unroll
      for (int i = 0; i < 8; i++) {
        int row = m0 + waveM + mt * 16 + ((lane < 16) ? i : i + 8);
        int col = n0 + waveN + nt * 16 + (lane & 15);
        float v = acc[mt][nt][i];
        if constexpr (OUT_F32) {
          ((float*)Cv)[(size_t)row * N + col] = v + bias[col];
        } else {
          ((bf16_t*)Cv)[(size_t)row * N + col] = to_bf16(v);
        }
      }
}

// ---------------- flash attention: B=2,S=4096,L=1024,H=16,D=64 ----------------
__global__ __launch_bounds__(128) void attn_kernel(
    const bf16_t* __restrict__ Q, const bf16_t* __restrict__ Kv,
    const bf16_t* __restrict__ V, bf16_t* __restrict__ ctx) {
  __shared__ bf16_t vScr[64 * 64];      // V^T tile [d][l], shared by all 4 waves
  __shared__ bf16_t pScr[4][16 * 64];   // per-wave P tile [m][l]
  const int tid = threadIdx.x, lane = tid & 31, wid = tid >> 5;
  int bid = blockIdx.x;
  const int nQB = 4096 / 64;
  int qb = bid % nQB; bid /= nQB;
  int h = bid % 16;   bid /= 16;
  int b = bid;
  const int q0 = qb * 64 + wid * 16;

  // Q fragments for K-chunks d0=0 and d0=32 (loaded once)
  BF16x16 qF[2];
  {
    const int rowq = b * 4096 + q0 + (lane & 15);
    const int kb0 = (lane < 16) ? 0 : 8;
    const bf16_t* p = Q + (size_t)rowq * 1024 + h * 64 + kb0;
#pragma unroll
    for (int d = 0; d < 2; d++) {
      qF[d].h[0] = *(const v8bf*)(p + d * 32);
      qF[d].h[1] = *(const v8bf*)(p + d * 32 + 16);
    }
  }

  const v8f zf = {0.f, 0.f, 0.f, 0.f, 0.f, 0.f, 0.f, 0.f};
  v8f accO[4];
#pragma unroll
  for (int i = 0; i < 4; i++) accO[i] = zf;
  float runMax[8], runSum[8];
#pragma unroll
  for (int i = 0; i < 8; i++) { runMax[i] = -__builtin_inff(); runSum[i] = 0.f; }

  const bf16_t* Kbase = Kv + (size_t)(b * 1024) * 1024 + h * 64;
  const bf16_t* Vbase = V + (size_t)(b * 1024) * 1024 + h * 64;
  bf16_t* ps = pScr[wid];

  for (int l0 = 0; l0 < 1024; l0 += 64) {
    // cooperatively stage V^T: thread -> (l row, 32-wide d half), coalesced reads, b16 scatter
    {
      const int lrel = tid >> 1, dh = (tid & 1) * 32;
      const bf16_t* vp = Vbase + (size_t)(l0 + lrel) * 1024 + dh;
      v8bf t0 = *(const v8bf*)(vp);
      v8bf t1 = *(const v8bf*)(vp + 8);
      v8bf t2 = *(const v8bf*)(vp + 16);
      v8bf t3 = *(const v8bf*)(vp + 24);
#pragma unroll
      for (int j = 0; j < 8; j++) {
        vScr[(dh + j) * 64 + lrel] = t0[j];
        vScr[(dh + j + 8) * 64 + lrel] = t1[j];
        vScr[(dh + j + 16) * 64 + lrel] = t2[j];
        vScr[(dh + j + 24) * 64 + lrel] = t3[j];
      }
    }
    __syncthreads();

    // scores: Q (16x64) @ K^T chunk (64 x 64) -> 4 tiles; K^T columns are contiguous K rows
    v8f sc[4];
#pragma unroll
    for (int i = 0; i < 4; i++) sc[i] = zf;
    const int kofs = (lane < 16) ? 0 : 8;
#pragma unroll
    for (int nt = 0; nt < 4; nt++) {
      const bf16_t* kp = Kbase + (size_t)(l0 + nt * 16 + (lane & 15)) * 1024 + kofs;
#pragma unroll
      for (int d = 0; d < 2; d++) {
        BF16x16 bF;
        bF.h[0] = *(const v8bf*)(kp + d * 32);
        bF.h[1] = *(const v8bf*)(kp + d * 32 + 16);
        sc[nt] = wmma_bf16(qF[d], bF, sc[nt]);
      }
    }
#pragma unroll
    for (int nt = 0; nt < 4; nt++)
#pragma unroll
      for (int i = 0; i < 8; i++) sc[nt][i] *= 0.125f;  // D^-0.5

    // online softmax: rows of a C tile live in one 16-lane half -> xor-shuffle reduce
#pragma unroll
    for (int i = 0; i < 8; i++) {
      float m = sc[0][i];
      m = fmaxf(m, sc[1][i]); m = fmaxf(m, sc[2][i]); m = fmaxf(m, sc[3][i]);
#pragma unroll
      for (int x = 1; x < 16; x <<= 1) m = fmaxf(m, __shfl_xor(m, x, 32));
      float nm = fmaxf(runMax[i], m);
      float corr = __expf(runMax[i] - nm);
      runMax[i] = nm;
      runSum[i] *= corr;
#pragma unroll
      for (int nt = 0; nt < 4; nt++) accO[nt][i] *= corr;
      float rs = 0.f;
#pragma unroll
      for (int nt = 0; nt < 4; nt++) {
        float p = __expf(sc[nt][i] - nm);
        sc[nt][i] = p;
        rs += p;
      }
#pragma unroll
      for (int x = 1; x < 16; x <<= 1) rs += __shfl_xor(rs, x, 32);
      runSum[i] += rs;
    }

    // P (C layout) -> LDS bf16 [m][l] so it can be re-read in A-fragment layout
#pragma unroll
    for (int nt = 0; nt < 4; nt++)
#pragma unroll
      for (int i = 0; i < 8; i++) {
        int row = (lane < 16) ? i : i + 8;
        ps[row * 64 + nt * 16 + (lane & 15)] = to_bf16(sc[nt][i]);
      }

    // O += P @ V : K dim = l (2 chunks of 32), N dim = d (4 tiles)
    const int kb0 = (lane < 16) ? 0 : 8;
    const int ks0 = (lane < 16) ? 0 : 8;
#pragma unroll
    for (int kc = 0; kc < 64; kc += 32) {
      BF16x16 aP;
      const bf16_t* pp = ps + (lane & 15) * 64 + kc + kb0;
      aP.h[0] = *(const v8bf*)pp;
      aP.h[1] = *(const v8bf*)(pp + 16);
#pragma unroll
      for (int nt = 0; nt < 4; nt++) {
        BF16x16 bV;
        const bf16_t* vp = &vScr[(nt * 16 + (lane & 15)) * 64 + kc + ks0];
        bV.h[0] = *(const v8bf*)vp;
        bV.h[1] = *(const v8bf*)(vp + 16);
        accO[nt] = wmma_bf16(aP, bV, accO[nt]);
      }
    }
    __syncthreads();
  }

  // normalize and write context (bf16, feeds output-projection GEMM)
#pragma unroll
  for (int nt = 0; nt < 4; nt++)
#pragma unroll
    for (int i = 0; i < 8; i++) {
      int row = q0 + ((lane < 16) ? i : i + 8);
      int col = h * 64 + nt * 16 + (lane & 15);
      float v = accO[nt][i] / runSum[i];
      ctx[(size_t)(b * 4096 + row) * 1024 + col] = to_bf16(v);
    }
}

// ---------------- host launcher ----------------
extern "C" void kernel_launch(void* const* d_in, const int* in_sizes, int n_in,
                              void* d_out, int out_size, void* d_ws, size_t ws_size,
                              hipStream_t stream) {
  (void)in_sizes; (void)n_in; (void)out_size; (void)ws_size;
  const float* hs  = (const float*)d_in[0];
  const float* enc = (const float*)d_in[1];
  const float* wq  = (const float*)d_in[2];
  const float* wk  = (const float*)d_in[3];
  const float* wv  = (const float*)d_in[4];
  const float* wo  = (const float*)d_in[5];
  const float* bo  = (const float*)d_in[6];
  const float* qd  = (const float*)d_in[7];
  const float* qu  = (const float*)d_in[8];
  const float* kd  = (const float*)d_in[9];
  const float* ku  = (const float*)d_in[10];
  const float* vd  = (const float*)d_in[11];
  const float* vu  = (const float*)d_in[12];
  const float* od  = (const float*)d_in[13];
  const float* ou  = (const float*)d_in[14];
  float* out = (float*)d_out;

  const int MS = 2 * 4096;  // B*S
  const int ML = 2 * 1024;  // B*L

  char* base = (char*)d_ws;
  size_t off = 0;
  auto alloc = [&](size_t bytes) -> void* {
    void* p = base + off;
    off += (bytes + 255) & ~(size_t)255;
    return p;
  };
  bf16_t* hsB  = (bf16_t*)alloc((size_t)MS * 1024 * 2);
  bf16_t* encB = (bf16_t*)alloc((size_t)ML * 768 * 2);
  bf16_t* wqT  = (bf16_t*)alloc((size_t)1024 * 1024 * 2);
  bf16_t* wkT  = (bf16_t*)alloc((size_t)1024 * 768 * 2);
  bf16_t* wvT  = (bf16_t*)alloc((size_t)1024 * 768 * 2);
  bf16_t* woT  = (bf16_t*)alloc((size_t)1024 * 1024 * 2);
  bf16_t* Qb   = (bf16_t*)alloc((size_t)MS * 1024 * 2);
  bf16_t* Kb   = (bf16_t*)alloc((size_t)ML * 1024 * 2);
  bf16_t* Vb   = (bf16_t*)alloc((size_t)ML * 1024 * 2);
  bf16_t* ctxB = (bf16_t*)alloc((size_t)MS * 1024 * 2);

  // activations to bf16
  cvt_f32_bf16<<<(MS * 1024 / 4 + 255) / 256, 256, 0, stream>>>(hs, hsB, MS * 1024 / 4);
  cvt_f32_bf16<<<(ML * 768 / 4 + 255) / 256, 256, 0, stream>>>(enc, encB, ML * 768 / 4);

  // fold LoRA into weights, transpose, to bf16
  fold_w<<<(1024 * 1024 + 255) / 256, 256, 0, stream>>>(wq, qd, qu, wqT, 1024, 1024);
  fold_w<<<(768 * 1024 + 255) / 256, 256, 0, stream>>>(wk, kd, ku, wkT, 768, 1024);
  fold_w<<<(768 * 1024 + 255) / 256, 256, 0, stream>>>(wv, vd, vu, wvT, 768, 1024);
  fold_w<<<(1024 * 1024 + 255) / 256, 256, 0, stream>>>(wo, od, ou, woT, 1024, 1024);

  // Q/K/V projections (WMMA GEMMs, bf16 out)
  gemm_bf16<false><<<dim3(1024 / 128, MS / 128), 256, 0, stream>>>(hsB, wqT, nullptr, Qb, MS, 1024, 1024);
  gemm_bf16<false><<<dim3(1024 / 128, ML / 128), 256, 0, stream>>>(encB, wkT, nullptr, Kb, ML, 1024, 768);
  gemm_bf16<false><<<dim3(1024 / 128, ML / 128), 256, 0, stream>>>(encB, wvT, nullptr, Vb, ML, 1024, 768);

  // flash attention
  attn_kernel<<<2 * 16 * (4096 / 64), 128, 0, stream>>>(Qb, Kb, Vb, ctxB);

  // output projection + bias (f32 out)
  gemm_bf16<true><<<dim3(1024 / 128, MS / 128), 256, 0, stream>>>(ctxB, woT, bo, out, MS, 1024, 1024);
}
